// SpinConvNet_48473000903104
// MI455X (gfx1250) — matrix operand order repaired and checked by reading, not compiled, not run on hardware
//
#include <hip/hip_runtime.h>
#include <hip/hip_bf16.h>
#include <cstdint>
#include <cstddef>

// ---------------------------------------------------------------------------
// SpinConv-style GNN for MI455X (gfx1250, wave32, WMMA).
// Dominant cost: per-round GEMM (E=20480) x (3072) x (768) done with
// v_wmma_f32_16x16x32_bf16; A = gathered per-node sphere grids (bf16),
// B = pre-rolled spin weights Wk (bf16, 4.7MB, L2-resident).
// M-tile = 32 edges per block so each B fragment feeds two WMMAs.
// ---------------------------------------------------------------------------

#define ATOMT 10
#define PHI_N 8
#define THETA_N 12
#define MDIM 32
#define DDIM 64
#define NV 2048
#define NC 32
#define NE 20480
#define PTM 3072          // PHI*THETA*M
#define NOUT 768          // THETA*D
#define PI_F 3.14159265358979323846f

typedef __bf16 bf16_t;
typedef __attribute__((ext_vector_type(16))) __bf16 v16bf;
typedef __attribute__((ext_vector_type(8)))  float  v8f;
typedef __attribute__((ext_vector_type(4)))  unsigned int u32x4;

union FragAB { v16bf v; u32x4 q[2]; };

__device__ __forceinline__ float silu_f(float v) { return v / (1.f + __expf(-v)); }

// ---------------------------------------------------------------------------
// Generic SpinConv embedding block up to the B-mixture stage.
// x (LDS, IN floats) holds the input; on return x holds the mixed vector.
// tmp (LDS, IN floats) is scratch. One wave processes one sample; lane-lockstep
// LDS RAW within a wave is handled by compiler-inserted s_wait_dscnt.
// ---------------------------------------------------------------------------
template <int IN>
__device__ __forceinline__ void emb_mixed(
    const float* __restrict__ W1, const float* __restrict__ b1,
    const float* __restrict__ W2, const float* __restrict__ b2,
    const float* __restrict__ attW, const float* __restrict__ attB,
    int t0, int t1, float* x, float* tmp, int lane)
{
  // type-conditioned softmax mixture weights (8 branches)
  float wb[8];
  float mx = -1e30f;
  #pragma unroll
  for (int b = 0; b < 8; ++b) {
    float l = attB[b] + attW[t0 * 8 + b];
    if (t1 >= 0) l += attW[t1 * 8 + b];
    wb[b] = l; mx = fmaxf(mx, l);
  }
  float s = 0.f;
  #pragma unroll
  for (int b = 0; b < 8; ++b) { wb[b] = __expf(wb[b] - mx); s += wb[b]; }
  float invs = 1.f / s;
  #pragma unroll
  for (int b = 0; b < 8; ++b) wb[b] *= invs;

  constexpr int PER = IN / 32;
  // fc1 + silu -> tmp
  float h[PER];
  #pragma unroll
  for (int r = 0; r < PER; ++r) {
    int j = lane + 32 * r;
    float a = b1[j];
    for (int i = 0; i < IN; ++i) a += x[i] * W1[i * IN + j];
    h[r] = silu_f(a);
  }
  #pragma unroll
  for (int r = 0; r < PER; ++r) tmp[lane + 32 * r] = h[r];

  // fc2 + silu, weighted mixture over 8 branches -> back into x
  float m[PER];
  #pragma unroll
  for (int r = 0; r < PER; ++r) {
    int i = lane + 32 * r;
    float acc = 0.f;
    for (int b = 0; b < 8; ++b) {
      float a = b2[b * IN + i];
      for (int j = 0; j < IN; ++j) a += tmp[j] * W2[j * (8 * IN) + b * IN + i];
      acc += wb[b] * silu_f(a);
    }
    m[r] = acc;
  }
  #pragma unroll
  for (int r = 0; r < PER; ++r) x[lane + 32 * r] = m[r];
}

// ---------------------------------------------------------------------------
// d_out zero (harness poisons once; we must re-zero each call before atomics)
// ---------------------------------------------------------------------------
__global__ void zero_out_kernel(float* o, int n)
{
  int i = blockIdx.x * 256 + threadIdx.x;
  if (i < n) o[i] = 0.f;
}

// ---------------------------------------------------------------------------
// Build rolled spin weights in bf16:  Bm[n][kk], n = kroll*64 + d,
// kk = (p*THETA + t)*M + m,  value = spin.w[(p*THETA + (t+kroll)%THETA)*M + m][d]
// ---------------------------------------------------------------------------
__global__ __launch_bounds__(256) void build_wk_kernel(
    const float* __restrict__ spinw, bf16_t* __restrict__ Bm)
{
  int idx = blockIdx.x * 256 + threadIdx.x;
  if (idx >= NOUT * PTM) return;
  int n = idx / PTM, kk = idx % PTM;
  int kroll = n >> 6, d = n & 63;
  int p = kk / (THETA_N * MDIM);
  int rem = kk % (THETA_N * MDIM);
  int t = rem / MDIM, m = rem % MDIM;
  int ts = t + kroll; if (ts >= THETA_N) ts -= THETA_N;
  float w = spinw[(size_t)(((p * THETA_N + ts) * MDIM) + m) * DDIM + d];
  Bm[(size_t)n * PTM + kk] = (bf16_t)w;
}

// ---------------------------------------------------------------------------
// Per-edge preprocessing: one wave per edge.
// Computes sphere bin, Gaussian distance basis, distance FC, dr_init/dr_msg,
// and the initial message embedding block.
// ---------------------------------------------------------------------------
__global__ __launch_bounds__(256) void prep_edges_kernel(
    const float* __restrict__ edge, const int* __restrict__ stype,
    const int* __restrict__ ttype,
    const float* __restrict__ dscale, const float* __restrict__ doffset,
    const float* __restrict__ dfw, const float* __restrict__ dfb,
    const float* __restrict__ ifw, const float* __restrict__ ifb,
    const float* __restrict__ mfw, const float* __restrict__ mfb,
    const float* __restrict__ e1w, const float* __restrict__ e1b,
    const float* __restrict__ e2w, const float* __restrict__ e2b,
    const float* __restrict__ e3w, const float* __restrict__ e3b,
    const float* __restrict__ eaw, const float* __restrict__ eab,
    int* __restrict__ bins, float* __restrict__ drmsg,
    float* __restrict__ message)
{
  __shared__ float xbuf[8][64];
  __shared__ float tbuf[8][64];
  int warp = threadIdx.x >> 5, lane = threadIdx.x & 31;
  int e = blockIdx.x * 8 + warp;
  float* x = xbuf[warp];
  float* tmp = tbuf[warp];

  float ex = edge[e * 3 + 0], ey = edge[e * 3 + 1], ez = edge[e * 3 + 2];
  float dist = sqrtf(ex * ex + ey * ey + ez * ez);
  float inv = 1.f / (dist + 1e-12f);
  float ux = ex * inv, uy = ey * inv, uz = ez * inv;
  float z = fminf(1.f, fmaxf(-1.f, uz));
  int pb = (int)(acosf(z) * (PHI_N / PI_F));
  pb = pb < 0 ? 0 : (pb > PHI_N - 1 ? PHI_N - 1 : pb);
  float tt = (atan2f(uy, ux) + PI_F) * (THETA_N / (2.f * PI_F));
  int tb = (int)tt;
  tb = tb < 0 ? 0 : (tb > THETA_N - 1 ? THETA_N - 1 : tb);
  if (lane == 0) bins[e] = pb * THETA_N + tb;

  int st = stype[e], tp = ttype[e];
  int et = st + tp * ATOMT;
  float d = dist * dscale[et] + doffset[et];

  // Gaussian basis (64 centers, 0..6), 2 per lane -> x
  #pragma unroll
  for (int r = 0; r < 2; ++r) {
    int i = lane + 32 * r;
    float c = (6.f / 63.f) * (float)i;
    float df = d - c;
    x[i] = __expf(-df * df * 8.f);   // 1/(2*0.25^2) = 8
  }
  // distance FC: dr -> tmp
  #pragma unroll
  for (int r = 0; r < 2; ++r) {
    int j = lane + 32 * r;
    float a = dfb[j];
    for (int i = 0; i < 64; ++i) a += x[i] * dfw[i * 64 + j];
    tmp[j] = a;
  }
  // dr_init (keep) / dr_msg (store)
  float di[2], dm[2];
  #pragma unroll
  for (int r = 0; r < 2; ++r) {
    int j = lane + 32 * r;
    float a = ifb[j], b = mfb[j];
    for (int i = 0; i < 64; ++i) { float t2 = tmp[i]; a += t2 * ifw[i * 64 + j]; b += t2 * mfw[i * 64 + j]; }
    di[r] = a; dm[r] = b;
  }
  #pragma unroll
  for (int r = 0; r < 2; ++r) {
    int j = lane + 32 * r;
    drmsg[(size_t)e * 64 + j] = dm[r];
    x[j] = di[r];
  }
  // init embedding block -> message (E x 32)
  emb_mixed<64>(e1w, e1b, e2w, e2b, eaw, eab, st, ATOMT + tp, x, tmp, lane);
  {
    int o = lane;
    float a = e3b[o];
    for (int i = 0; i < 64; ++i) a += x[i] * e3w[i * 32 + o];
    message[(size_t)e * 32 + o] = a;
  }
}

// ---------------------------------------------------------------------------
// Scatter incoming-edge messages onto per-node sphere grid; store bf16.
// One workgroup per node; 12KB LDS grid with ds_add_f32 accumulation.
// ---------------------------------------------------------------------------
__global__ __launch_bounds__(256) void scatter_grid_kernel(
    const int* __restrict__ tei, const int* __restrict__ bins,
    const float* __restrict__ message, bf16_t* __restrict__ G)
{
  __shared__ float Gl[PTM];
  int v = blockIdx.x, tid = threadIdx.x;
  for (int i = tid; i < PTM; i += 256) Gl[i] = 0.f;
  __syncthreads();
  int m = tid & 31;
  int csub = tid >> 5;
  for (int c = csub; c < NC; c += 8) {
    int e = tei[v * NC + c];
    int bin = bins[e];
    atomicAdd(&Gl[bin * 32 + m], message[(size_t)e * 32 + m]);
  }
  __syncthreads();
  for (int i = tid; i < PTM; i += 256)
    G[(size_t)v * PTM + i] = (bf16_t)Gl[i];
}

// ---------------------------------------------------------------------------
// Spin-conv GEMM: per 32-edge tile, 12 waves (one per theta-roll), each wave
// computes 32x64 outputs via 8 x v_wmma_f32_16x16x32_bf16 per 32-wide K step
// (two 16-edge A tiles reuse every B fragment), then SiLU(+bias) and mean
// over the 12 rolls via LDS reduction.
// Fragment layouts follow CDNA5 ISA 7.12.2 (16-bit A 16x32, B 32x16).
// ---------------------------------------------------------------------------
__global__ __launch_bounds__(384) void spin_gemm_kernel(
    const bf16_t* __restrict__ G, const bf16_t* __restrict__ Bm,
    const int* __restrict__ esi, const float* __restrict__ spin_b,
    float* __restrict__ sc)
{
  __shared__ float red[32 * 64];
  const int tid = threadIdx.x;
  const int lane = tid & 31;
  const int kroll = tid >> 5;                 // 0..11
  const int tile = blockIdx.x;                // edge tile (32 edges)
  for (int i = tid; i < 32 * 64; i += 384) red[i] = 0.f;
  __syncthreads();

  const int nsub = lane & 15;
  const int grp = lane >> 4;
  const int node0 = esi[tile * 32 + nsub];
  const int node1 = esi[tile * 32 + 16 + nsub];
  const bf16_t* __restrict__ arow0 = G + (size_t)node0 * PTM;
  const bf16_t* __restrict__ arow1 = G + (size_t)node1 * PTM;
  const bf16_t* __restrict__ brow = Bm + (size_t)(kroll * 64 + nsub) * PTM;

  v8f zero8 = {0.f, 0.f, 0.f, 0.f, 0.f, 0.f, 0.f, 0.f};
  v8f acc[2][4];
  #pragma unroll
  for (int h = 0; h < 2; ++h)
    #pragma unroll
    for (int j = 0; j < 4; ++j) acc[h][j] = zero8;

  #pragma unroll 2
  for (int k0 = 0; k0 < PTM; k0 += 32) {
    // A 16x32: lane m=lane&15, K groups [grp*8,+8) and [16+grp*8,+8)
    FragAB a0, a1;
    a0.q[0] = *(const u32x4*)(arow0 + k0 + grp * 8);
    a0.q[1] = *(const u32x4*)(arow0 + k0 + 16 + grp * 8);
    a1.q[0] = *(const u32x4*)(arow1 + k0 + grp * 8);
    a1.q[1] = *(const u32x4*)(arow1 + k0 + 16 + grp * 8);
    // B 32x16: lane n=lane&15, K groups [grp*8,+8) and [16+grp*8,+8)
    const bf16_t* bp = brow + k0 + grp * 8;
    #pragma unroll
    for (int j = 0; j < 4; ++j) {
      FragAB b;
      b.q[0] = *(const u32x4*)(bp + j * (16 * PTM));
      b.q[1] = *(const u32x4*)(bp + j * (16 * PTM) + 16);
      acc[0][j] = __builtin_amdgcn_wmma_f32_16x16x32_bf16(false, a0.v, false, b.v, (short)0, acc[0][j], false, false);
      acc[1][j] = __builtin_amdgcn_wmma_f32_16x16x32_bf16(false, a1.v, false, b.v, (short)0, acc[1][j], false, false);
    }
  }

  // bias + silu, accumulate mean-over-rolls into LDS
  const int moff = (lane < 16) ? 0 : 8;       // C layout: vgpr r -> M=r (lanes 0-15) / r+8 (lanes 16-31)
  #pragma unroll
  for (int h = 0; h < 2; ++h) {
    #pragma unroll
    for (int j = 0; j < 4; ++j) {
      #pragma unroll
      for (int r = 0; r < 8; ++r) {
        int m = h * 16 + moff + r;
        int d = j * 16 + nsub;
        float v = acc[h][j][r] + spin_b[d];
        atomicAdd(&red[m * 64 + d], silu_f(v));
      }
    }
  }
  __syncthreads();
  for (int i = tid; i < 32 * 64; i += 384)
    sc[(size_t)(tile * 32 + (i >> 6)) * 64 + (i & 63)] = red[i] * (1.f / 12.f);
}

// ---------------------------------------------------------------------------
// Residual message update: res = m2(m1(sc) + dr_msg); message += res.
// One wave per edge.
// ---------------------------------------------------------------------------
__global__ __launch_bounds__(256) void msg_update_kernel(
    const float* __restrict__ sc, const float* __restrict__ drmsg,
    const int* __restrict__ stype, const int* __restrict__ ttype,
    const float* __restrict__ m1w1, const float* __restrict__ m1b1,
    const float* __restrict__ m1w2, const float* __restrict__ m1b2,
    const float* __restrict__ m1w3, const float* __restrict__ m1b3,
    const float* __restrict__ m1aw, const float* __restrict__ m1ab,
    const float* __restrict__ m2w1, const float* __restrict__ m2b1,
    const float* __restrict__ m2w2, const float* __restrict__ m2b2,
    const float* __restrict__ m2w3, const float* __restrict__ m2b3,
    const float* __restrict__ m2aw, const float* __restrict__ m2ab,
    float* __restrict__ message)
{
  __shared__ float xbuf[8][64];
  __shared__ float tbuf[8][64];
  int warp = threadIdx.x >> 5, lane = threadIdx.x & 31;
  int e = blockIdx.x * 8 + warp;
  float* x = xbuf[warp];
  float* tmp = tbuf[warp];

  #pragma unroll
  for (int r = 0; r < 2; ++r) { int j = lane + 32 * r; x[j] = sc[(size_t)e * 64 + j]; }
  int st = stype[e], tp = ttype[e];

  emb_mixed<64>(m1w1, m1b1, m1w2, m1b2, m1aw, m1ab, st, ATOMT + tp, x, tmp, lane);
  float y[2];
  #pragma unroll
  for (int r = 0; r < 2; ++r) {
    int o = lane + 32 * r;
    float a = m1b3[o];
    for (int i = 0; i < 64; ++i) a += x[i] * m1w3[i * 64 + o];
    y[r] = a + drmsg[(size_t)e * 64 + o];
  }
  #pragma unroll
  for (int r = 0; r < 2; ++r) x[lane + 32 * r] = y[r];

  emb_mixed<64>(m2w1, m2b1, m2w2, m2b2, m2aw, m2ab, st, ATOMT + tp, x, tmp, lane);
  {
    int o = lane;
    float a = m2b3[o];
    for (int i = 0; i < 64; ++i) a += x[i] * m2w3[i * 32 + o];
    message[(size_t)e * 32 + o] += a;
  }
}

// ---------------------------------------------------------------------------
// Node aggregation + energy head; global scalar via wave reduce + atomicAdd.
// One wave per node.
// ---------------------------------------------------------------------------
__global__ __launch_bounds__(256) void energy_kernel(
    const int* __restrict__ tei, const float* __restrict__ message,
    const int* __restrict__ ntype,
    const float* __restrict__ g1w, const float* __restrict__ g1b,
    const float* __restrict__ g2w, const float* __restrict__ g2b,
    const float* __restrict__ g3w, const float* __restrict__ g3b,
    const float* __restrict__ gaw, const float* __restrict__ gab,
    float* __restrict__ out)
{
  __shared__ float xbuf[8][32];
  __shared__ float tbuf[8][32];
  int warp = threadIdx.x >> 5, lane = threadIdx.x & 31;
  int v = blockIdx.x * 8 + warp;
  float* x = xbuf[warp];
  float* tmp = tbuf[warp];

  float a = 0.f;
  for (int c = 0; c < NC; ++c)
    a += message[(size_t)tei[v * NC + c] * 32 + lane];
  x[lane] = a;

  emb_mixed<32>(g1w, g1b, g2w, g2b, gaw, gab, ntype[v], -1, x, tmp, lane);

  float part = x[lane] * g3w[lane];            // fc3: (32 -> 1)
  #pragma unroll
  for (int off = 16; off > 0; off >>= 1) part += __shfl_down(part, off, 32);
  if (lane == 0) atomicAdd(out, part + g3b[0]);
}

// ---------------------------------------------------------------------------
// Host-side launch. Input index mapping assumes insertion-order flattening of
// setup_inputs() and the nested params dict (w before b per linear layer).
// ---------------------------------------------------------------------------
enum {
  IN_TEI = 0, IN_ESI, IN_EDGE, IN_STYPE, IN_TTYPE, IN_NTYPE,
  P_DIST_SCALE, P_DIST_OFFSET, P_DIST_FCW, P_DIST_FCB,
  P_INITFC_W, P_INITFC_B, P_MSGFC_W, P_MSGFC_B,
  P_IE_FC1W, P_IE_FC1B, P_IE_FC2W, P_IE_FC2B, P_IE_FC3W, P_IE_FC3B, P_IE_ATTW, P_IE_ATTB,
  P_M1_FC1W, P_M1_FC1B, P_M1_FC2W, P_M1_FC2B, P_M1_FC3W, P_M1_FC3B, P_M1_ATTW, P_M1_ATTB,
  P_M2_FC1W, P_M2_FC1B, P_M2_FC2W, P_M2_FC2B, P_M2_FC3W, P_M2_FC3B, P_M2_ATTW, P_M2_ATTB,
  P_EN_FC1W, P_EN_FC1B, P_EN_FC2W, P_EN_FC2B, P_EN_FC3W, P_EN_FC3B, P_EN_ATTW, P_EN_ATTB,
  P_SPIN_W, P_SPIN_B
};

extern "C" void kernel_launch(void* const* d_in, const int* in_sizes, int n_in,
                              void* d_out, int out_size, void* d_ws, size_t ws_size,
                              hipStream_t stream)
{
  (void)in_sizes; (void)n_in; (void)ws_size;
  const int* tei   = (const int*)d_in[IN_TEI];
  const int* esi   = (const int*)d_in[IN_ESI];
  const float* edg = (const float*)d_in[IN_EDGE];
  const int* stype = (const int*)d_in[IN_STYPE];
  const int* ttype = (const int*)d_in[IN_TTYPE];
  const int* ntype = (const int*)d_in[IN_NTYPE];
  #define PP(i) ((const float*)d_in[(i)])

  // workspace carving (all fully re-initialized each call)
  char* ws = (char*)d_ws;
  size_t o = 0;
  auto take = [&](size_t nbytes) -> void* {
    void* p = (void*)(ws + o);
    o += (nbytes + 255) & ~(size_t)255;
    return p;
  };
  float*  message = (float*)take((size_t)NE * MDIM * sizeof(float));
  float*  drmsg   = (float*)take((size_t)NE * DDIM * sizeof(float));
  float*  sc      = (float*)take((size_t)NE * DDIM * sizeof(float));
  int*    bins    = (int*)take((size_t)NE * sizeof(int));
  bf16_t* G       = (bf16_t*)take((size_t)NV * PTM * sizeof(bf16_t));
  bf16_t* Bm      = (bf16_t*)take((size_t)NOUT * PTM * sizeof(bf16_t));

  zero_out_kernel<<<(out_size + 255) / 256, 256, 0, stream>>>((float*)d_out, out_size);
  build_wk_kernel<<<(NOUT * PTM + 255) / 256, 256, 0, stream>>>(PP(P_SPIN_W), Bm);

  prep_edges_kernel<<<NE / 8, 256, 0, stream>>>(
      edg, stype, ttype,
      PP(P_DIST_SCALE), PP(P_DIST_OFFSET), PP(P_DIST_FCW), PP(P_DIST_FCB),
      PP(P_INITFC_W), PP(P_INITFC_B), PP(P_MSGFC_W), PP(P_MSGFC_B),
      PP(P_IE_FC1W), PP(P_IE_FC1B), PP(P_IE_FC2W), PP(P_IE_FC2B),
      PP(P_IE_FC3W), PP(P_IE_FC3B), PP(P_IE_ATTW), PP(P_IE_ATTB),
      bins, drmsg, message);

  for (int round = 0; round < 2; ++round) {
    scatter_grid_kernel<<<NV, 256, 0, stream>>>(tei, bins, message, G);
    spin_gemm_kernel<<<NE / 32, 384, 0, stream>>>(G, Bm, esi, PP(P_SPIN_B), sc);
    msg_update_kernel<<<NE / 8, 256, 0, stream>>>(
        sc, drmsg, stype, ttype,
        PP(P_M1_FC1W), PP(P_M1_FC1B), PP(P_M1_FC2W), PP(P_M1_FC2B),
        PP(P_M1_FC3W), PP(P_M1_FC3B), PP(P_M1_ATTW), PP(P_M1_ATTB),
        PP(P_M2_FC1W), PP(P_M2_FC1B), PP(P_M2_FC2W), PP(P_M2_FC2B),
        PP(P_M2_FC3W), PP(P_M2_FC3B), PP(P_M2_ATTW), PP(P_M2_ATTB),
        message);
  }

  energy_kernel<<<NV / 8, 256, 0, stream>>>(
      tei, message, ntype,
      PP(P_EN_FC1W), PP(P_EN_FC1B), PP(P_EN_FC2W), PP(P_EN_FC2B),
      PP(P_EN_FC3W), PP(P_EN_FC3B), PP(P_EN_ATTW), PP(P_EN_ATTB),
      (float*)d_out);
  #undef PP
}
